// NeuralLongTermMemory_55422257988305
// MI455X (gfx1250) — matrix operand order repaired and checked by reading, not compiled, not run on hardware
//
#include <hip/hip_runtime.h>
#include <math.h>

// Problem constants (from reference)
#define Bz   8
#define Tz   128
#define DIN  128
#define DMEM 256
#define DOUT 128
#define CLIPV 10000.0f

typedef __attribute__((ext_vector_type(16))) _Float16 v16h;
typedef __attribute__((ext_vector_type(8)))  float    v8f;

// Fast sigmoid: v_exp_f32 + v_add_f32 + v_rcp_f32 (3 ops) instead of the
// ~10-op IEEE division sequence. Critical-path win for the serial scan.
__device__ __forceinline__ float sigmoidf_(float z) {
    return __builtin_amdgcn_rcpf(1.0f + __expf(-z));
}
__device__ __forceinline__ float clipf_(float v) { return fminf(fmaxf(v, -CLIPV), CLIPV); }

// ---------------------------------------------------------------------------
// Kernel 1: gate MLPs (theta/eta/alpha) for all B*T tokens.
//   hidden = silu(x @ w1g.T + b1g)   -> 1024 x 256 GEMM, K=128  (WMMA f16)
//   s      = hidden @ w2g.T + b2g    -> per-row reduction (LDS ds_add_f32)
//   gate   = softplus(s) (g==0) or sigmoid(s) (g==1,2)
// grid = (64 row-blocks of 16, 3 gates), block = 256 threads = 8 waves.
// Each wave owns 2 column tiles; the two WMMA accumulation chains are
// interleaved so the D->WMMA hazard slots of one chain are filled by the
// other (no v_nop stalls), and A is loaded once for both.
// ---------------------------------------------------------------------------
__global__ __launch_bounds__(256) void gates_kernel(
    const float* __restrict__ x,
    const float* __restrict__ tw1, const float* __restrict__ tb1,
    const float* __restrict__ tw2, const float* __restrict__ tb2,
    const float* __restrict__ ew1, const float* __restrict__ eb1,
    const float* __restrict__ ew2, const float* __restrict__ eb2,
    const float* __restrict__ aw1, const float* __restrict__ ab1,
    const float* __restrict__ aw2, const float* __restrict__ ab2,
    float* __restrict__ gtheta, float* __restrict__ geta, float* __restrict__ galpha)
{
    const int tid = threadIdx.x;
    const int g   = blockIdx.y;          // 0=theta 1=eta 2=alpha
    const int r0  = blockIdx.x * 16;     // first of 16 token rows

    const float* w1g = (g == 0) ? tw1 : (g == 1) ? ew1 : aw1;
    const float* b1g = (g == 0) ? tb1 : (g == 1) ? eb1 : ab1;
    const float* w2g = (g == 0) ? tw2 : (g == 1) ? ew2 : aw2;
    const float* b2g = (g == 0) ? tb2 : (g == 1) ? eb2 : ab2;
    float* gout      = (g == 0) ? gtheta : (g == 1) ? geta : galpha;

    __shared__ __align__(16) _Float16 sA[16 * DIN];      // x block  [m][k], f16
    __shared__ __align__(16) _Float16 sB[DMEM * DIN];    // w1g      [n][k], f16 (64KB)
    __shared__ float sBias[DMEM];
    __shared__ float sW2[DMEM];
    __shared__ float rowAcc[16];

    // Stage x block (16x128) as f16
    for (int e = tid; e < 16 * DIN; e += 256)
        sA[e] = (_Float16)x[(size_t)(r0) * DIN + e];
    // Stage gate weight (256x128) as f16
    for (int e = tid; e < DMEM * DIN; e += 256)
        sB[e] = (_Float16)w1g[e];
    if (tid < DMEM) { sBias[tid] = b1g[tid]; sW2[tid] = w2g[tid]; }
    if (tid < 16) rowAcc[tid] = 0.0f;
    __syncthreads();

    const int lane = tid & 31;
    const int wave = tid >> 5;
    const int hi   = lane >> 4;          // lane group (0: lanes 0-15, 1: lanes 16-31)
    const int mrow = lane & 15;          // M index for A, N index for B/C

    const int n0 = (wave * 2) * 16 + mrow;       // column of tile 0
    const int n1 = n0 + 16;                      // column of tile 1
    v8f c0 = {}, c1 = {};

    #pragma unroll
    for (int c4 = 0; c4 < 4; ++c4) {             // K chunks of 32
        const int kbase = c4 * 32;
        union { v16h v; unsigned int u[8]; } ua, ub0, ub1;
        // A, contiguous k order (same as B): lanes 0-15 hold K=0..15, lanes 16-31 K=16..31
        const unsigned int* aRow =
            (const unsigned int*)(sA + mrow * DIN + kbase);
        #pragma unroll
        for (int q = 0; q < 8; ++q) {
            ua.u[q] = aRow[q + 8 * hi];
        }
        // B, contiguous k order (same as A): lanes 0-15 hold K=0..15, lanes 16-31 K=16..31
        const unsigned int* bRow0 =
            (const unsigned int*)(sB + n0 * DIN + kbase + 16 * hi);
        const unsigned int* bRow1 =
            (const unsigned int*)(sB + n1 * DIN + kbase + 16 * hi);
        #pragma unroll
        for (int q = 0; q < 8; ++q) { ub0.u[q] = bRow0[q]; ub1.u[q] = bRow1[q]; }

        // Interleaved independent accumulation chains (hazard slots overlap)
        c0 = __builtin_amdgcn_wmma_f32_16x16x32_f16(
                 false, ua.v, false, ub0.v, (short)0, c0, false, false);
        c1 = __builtin_amdgcn_wmma_f32_16x16x32_f16(
                 false, ua.v, false, ub1.v, (short)0, c1, false, false);
    }

    // Epilogue: silu + second-layer partial reduction (both tiles hit the
    // same 16 rows -> one ds_add_f32 per row per lane)
    {
        const float bs0 = sBias[n0], w20 = sW2[n0];
        const float bs1 = sBias[n1], w21 = sW2[n1];
        #pragma unroll
        for (int j = 0; j < 8; ++j) {
            const int M = j + 8 * hi;                   // C layout row
            float z0 = c0[j] + bs0;
            float z1 = c1[j] + bs1;
            float p  = (z0 * sigmoidf_(z0)) * w20 + (z1 * sigmoidf_(z1)) * w21;
            atomicAdd(&rowAcc[M], p);                   // ds_add_f32
        }
    }
    __syncthreads();

    if (tid < 16) {
        float s = rowAcc[tid] + b2g[0];
        float v;
        if (g == 0) { // softplus, numerically stable
            v = (s > 20.0f) ? s : __logf(1.0f + __expf(s));
        } else {
            v = sigmoidf_(s);
        }
        gout[r0 + tid] = v;
    }
}

// ---------------------------------------------------------------------------
// Kernel 2: sequential fast-weight scan. One workgroup per batch (8 WGPs).
// State placement:
//   W1, M1w  -> LDS (128x256 f32 each, row stride 261 -> conflict-free both ways)
//   W0, M0w  -> global workspace (L2-resident), float4 coalesced
// 256 threads, T=128 dependent steps with workgroup barriers between phases.
// Token staging is double-buffered via GLOBAL_LOAD_ASYNC_TO_LDS_B128: the
// async load for step t+1 is issued at the top of step t, so its L2 latency
// hides behind the entire step body (ASYNCcnt tracks it; s_wait_asynccnt 0
// at the next step's start). Gate scalars are software-pipelined likewise.
// ---------------------------------------------------------------------------
#define STR1 261   // 261 mod 64 = 5, gcd(5,64)=1: row AND column reads conflict-free

__global__ __launch_bounds__(256) void scan_kernel(
    const float* __restrict__ x,
    const float* __restrict__ w0, const float* __restrict__ b0,
    const float* __restrict__ w1, const float* __restrict__ b1,
    const float* __restrict__ gtheta, const float* __restrict__ geta,
    const float* __restrict__ galpha,
    float* __restrict__ W0g, float* __restrict__ M0g,
    float* __restrict__ y)
{
    const int tid = threadIdx.x;
    const int b   = blockIdx.x;

    __shared__ __align__(16) float lW1[128 * STR1];   // ~130.5 KB
    __shared__ __align__(16) float lM1[128 * STR1];   // ~130.5 KB
    __shared__ __align__(16) float lB0[DMEM], lM0b[DMEM];
    __shared__ __align__(16) float lB1[DOUT], lM1b[DOUT];
    __shared__ __align__(16) float sxt[2 * DIN];      // double-buffered token
    __shared__ __align__(16) float sz0[DMEM], sh0[DMEM], sd0[DMEM];
    __shared__ __align__(16) float sd1[DOUT];
    __shared__ __align__(16) float spart[256];

    float* W0b = W0g + (size_t)b * (DMEM * DIN);
    float* M0b = M0g + (size_t)b * (DMEM * DIN);

    // ---- init state (deterministic every call) ----
    for (int e = tid; e < DMEM * DIN; e += 256) { W0b[e] = w0[e]; M0b[e] = 0.0f; }
    for (int e = tid; e < DOUT * DMEM; e += 256) {
        int o = e >> 8, m = e & 255;
        lW1[o * STR1 + m] = w1[e];
        lM1[o * STR1 + m] = 0.0f;
    }
    lB0[tid] = b0[tid]; lM0b[tid] = 0.0f;
    if (tid < DOUT) { lB1[tid] = b1[tid]; lM1b[tid] = 0.0f; }
    __threadfence_block();
    __syncthreads();

    // issue async token load for t=0 into buffer 0 (wave 0, 16B/lane = 512B)
    if (tid < 32) {
        const unsigned long long gaddr =
            (unsigned long long)(&x[((size_t)(b * Tz)) * DIN + tid * 4]);
        const unsigned ldsa = (unsigned)(uintptr_t)(&sxt[tid * 4]);
        asm volatile("global_load_async_to_lds_b128 %0, %1, off"
                     :: "v"(ldsa), "v"(gaddr) : "memory");
    }

    // software-pipelined gate scalars
    float thc = gtheta[b * Tz], etc_ = geta[b * Tz], alc = galpha[b * Tz];

    for (int t = 0; t < Tz; ++t) {
        // prefetch next step's gate scalars (consumed next iteration)
        float thn = 0.0f, etn = 0.0f, aln = 0.0f;
        if (t + 1 < Tz) {
            thn = gtheta[b * Tz + t + 1];
            etn = geta  [b * Tz + t + 1];
            aln = galpha[b * Tz + t + 1];
        }
        const float th = thc, et = etc_, keep = 1.0f - alc;
        const float* xbuf = sxt + (t & 1) * DIN;
        const float4* xt4 = (const float4*)xbuf;

        // ---- wait for token t, then launch async load of token t+1 into
        //      the other buffer (hides behind the whole step body) ----
        if (tid < 32) {
            asm volatile("s_wait_asynccnt 0x0" ::: "memory");
            if (t + 1 < Tz) {
                const unsigned long long gaddr =
                    (unsigned long long)(&x[((size_t)(b * Tz + t + 1)) * DIN + tid * 4]);
                const unsigned ldsa =
                    (unsigned)(uintptr_t)(&sxt[((t + 1) & 1) * DIN + tid * 4]);
                asm volatile("global_load_async_to_lds_b128 %0, %1, off"
                             :: "v"(ldsa), "v"(gaddr) : "memory");
            }
        }
        __syncthreads();

        // ---- phase 1: z0 = W0 xt + B0 ; h0 = silu(z0) (thread = row m) ----
        {
            const int m = tid;
            const float4* row = (const float4*)(W0b + m * DIN);
            float acc = lB0[m];
            #pragma unroll 8
            for (int i4 = 0; i4 < DIN / 4; ++i4) {
                float4 w = row[i4], xv = xt4[i4];
                acc += w.x * xv.x + w.y * xv.y + w.z * xv.z + w.w * xv.w;
            }
            sz0[m] = acc;
            sh0[m] = acc * sigmoidf_(acc);
        }
        __syncthreads();

        // ---- phase 2: z1 = W1 h0 + B1 (2 threads per output row) ----
        {
            const int o = tid & 127, half = tid >> 7, base = half * 128;
            float acc = 0.0f;
            #pragma unroll 8
            for (int mm = 0; mm < 128; ++mm)
                acc += lW1[o * STR1 + base + mm] * sh0[base + mm];
            spart[tid] = acc;
        }
        __syncthreads();
        if (tid < DOUT) {
            float z1 = spart[tid] + spart[tid + 128] + lB1[tid];
            y[((size_t)(b * Tz + t)) * DOUT + tid] = z1;
            sd1[tid] = 2.0f * (z1 - xbuf[tid]);
        }
        __syncthreads();

        // ---- phase 3: d0 = (W1^T d1) * silu'(z0) ; bias updates ----
        {
            const int m = tid;
            float acc = 0.0f;
            #pragma unroll 8
            for (int o = 0; o < DOUT; ++o)
                acc += lW1[o * STR1 + m] * sd1[o];   // conflict-free column read
            float z0 = sz0[m], s = sigmoidf_(z0);
            float d0 = acc * (s * (1.0f + z0 * (1.0f - s)));
            sd0[m] = d0;
            float db0 = clipf_(d0);
            float m0  = et * lM0b[m] - th * db0;
            lM0b[m] = m0;
            lB0[m]  = keep * lB0[m] + m0;
            if (tid < DOUT) {
                float db1 = clipf_(sd1[tid]);
                float m1  = et * lM1b[tid] - th * db1;
                lM1b[tid] = m1;
                lB1[tid]  = keep * lB1[tid] + m1;
            }
        }
        __syncthreads();

        // ---- phase 4a: W1/M1w update in LDS (elem (o=j, m=tid)) ----
        {
            const float h0m = sh0[tid & 255];
            #pragma unroll 4
            for (int j = 0; j < DOUT; ++j) {
                const int idx = j * STR1 + tid;
                float gg = clipf_(sd1[j] * h0m);
                float mv = et * lM1[idx] - th * gg;
                lM1[idx] = mv;
                lW1[idx] = keep * lW1[idx] + mv;
            }
        }
        // ---- phase 4b: W0/M0w update in global, float4 coalesced ----
        {
            float4* W04 = (float4*)W0b;
            float4* M04 = (float4*)M0b;
            #pragma unroll 4
            for (int j = 0; j < 32; ++j) {
                const int idx4 = tid + 256 * j;      // 8192 float4 total
                const int m    = idx4 >> 5;          // row 0..255
                const int ib   = idx4 & 31;          // float4 col 0..31
                float  d0v = sd0[m];
                float4 xv  = xt4[ib];
                float4 mo  = M04[idx4];
                float4 wo  = W04[idx4];
                float4 mv, wv;
                mv.x = et * mo.x - th * clipf_(d0v * xv.x);
                mv.y = et * mo.y - th * clipf_(d0v * xv.y);
                mv.z = et * mo.z - th * clipf_(d0v * xv.z);
                mv.w = et * mo.w - th * clipf_(d0v * xv.w);
                wv.x = keep * wo.x + mv.x;
                wv.y = keep * wo.y + mv.y;
                wv.z = keep * wo.z + mv.z;
                wv.w = keep * wo.w + mv.w;
                M04[idx4] = mv;
                W04[idx4] = wv;
            }
        }
        __threadfence_block();
        __syncthreads();

        thc = thn; etc_ = etn; alc = aln;
    }
}

// ---------------------------------------------------------------------------
// Launch. Inputs (setup_inputs order):
//  0:x 1:w0 2:b0 3:w1 4:b1 5:tw1 6:tb1 7:tw2 8:tb2 9:ew1 10:eb1 11:ew2 12:eb2
//  13:aw1 14:ab1 15:aw2 16:ab2
// Workspace: theta/eta/alpha (3*1024 f32) + W0 state (8*32768) + M0w (8*32768)
//   = 527360 floats ~ 2.06 MB.
// ---------------------------------------------------------------------------
extern "C" void kernel_launch(void* const* d_in, const int* in_sizes, int n_in,
                              void* d_out, int out_size, void* d_ws, size_t ws_size,
                              hipStream_t stream) {
    const float* x   = (const float*)d_in[0];
    const float* w0  = (const float*)d_in[1];
    const float* b0  = (const float*)d_in[2];
    const float* w1  = (const float*)d_in[3];
    const float* b1  = (const float*)d_in[4];
    const float* tw1 = (const float*)d_in[5];
    const float* tb1 = (const float*)d_in[6];
    const float* tw2 = (const float*)d_in[7];
    const float* tb2 = (const float*)d_in[8];
    const float* ew1 = (const float*)d_in[9];
    const float* eb1 = (const float*)d_in[10];
    const float* ew2 = (const float*)d_in[11];
    const float* eb2 = (const float*)d_in[12];
    const float* aw1 = (const float*)d_in[13];
    const float* ab1 = (const float*)d_in[14];
    const float* aw2 = (const float*)d_in[15];
    const float* ab2 = (const float*)d_in[16];

    float* wsf    = (float*)d_ws;
    float* gtheta = wsf;
    float* geta   = wsf + Bz * Tz;
    float* galpha = wsf + 2 * Bz * Tz;
    float* W0g    = wsf + 3 * Bz * Tz;
    float* M0g    = W0g + (size_t)Bz * DMEM * DIN;

    gates_kernel<<<dim3(64, 3), 256, 0, stream>>>(
        x, tw1, tb1, tw2, tb2, ew1, eb1, ew2, eb2, aw1, ab1, aw2, ab2,
        gtheta, geta, galpha);

    scan_kernel<<<dim3(Bz), 256, 0, stream>>>(
        x, w0, b0, w1, b1, gtheta, geta, galpha, W0g, M0g, (float*)d_out);
}
